// Self_Attention_12154757448229
// MI455X (gfx1250) — matrix operand order repaired and checked by reading, not compiled, not run on hardware
//
#include <hip/hip_runtime.h>
#include <hip/hip_bf16.h>
#include <math.h>

typedef __attribute__((ext_vector_type(16))) _Float16 v16h;
typedef __attribute__((ext_vector_type(8)))  _Float16 h8;
typedef __attribute__((ext_vector_type(8)))  float    v8f;

#define S_LEN 2048
#define D_DIM 128
#define QTILE 128              // query rows per block (8 waves x 16)
#define KT    64               // keys per iteration
#define NKT   (S_LEN / KT)
#define QK_SCALE 0.08838834764831845f  // 1/sqrt(128)
#define NEG_BIG (-1.0e9f)

__global__ __launch_bounds__(256) void fattn_f16wmma_kernel(
    const float* __restrict__ Q,
    const float* __restrict__ K,
    const float* __restrict__ V,
    float* __restrict__ O)
{
    // K tile: [key][d], row stride 136 halves = 272B (16B aligned, 68-dword = 4-bank row shift)
    __shared__ __align__(16) _Float16 kT[KT][136];
    // V tile transposed: [d][key], row stride 72 halves = 144B (16B aligned, 36-dword row shift)
    __shared__ __align__(16) _Float16 vT[D_DIM][72];
    // per-wave P scratch, [m][key_local], row stride 72 halves
    __shared__ __align__(16) _Float16 pS[8][16][72];

    const int tid  = threadIdx.x;
    const int wave = tid >> 5;
    const int lane = tid & 31;
    const int lo   = lane & 15;   // N / M position within 16
    const int hi   = lane >> 4;   // half-wave selector

    const int bh    = blockIdx.y;
    const int qBase = blockIdx.x * QTILE + wave * 16;

    // ---- load this wave's 16x128 Q tile into WMMA A layout, pre-scaled, f16 ----
    // A layout (16-bit A 16x32): lane m = lo; halves 0..7 -> k = 8*hi + j,
    // halves 8..15 -> k = 16 + 8*hi + j  (within each 32-wide d chunk).
    const float* qrow = Q + ((size_t)bh * S_LEN + qBase + lo) * D_DIM;
    v16h qa[4];
#pragma unroll
    for (int dc = 0; dc < 4; ++dc) {
        const int b1 = dc * 32 + 8 * hi;
        const int b2 = dc * 32 + 16 + 8 * hi;
#pragma unroll
        for (int j = 0; j < 8; ++j) {
            qa[dc][j]     = (_Float16)(qrow[b1 + j] * QK_SCALE);
            qa[dc][8 + j] = (_Float16)(qrow[b2 + j] * QK_SCALE);
        }
    }

    // Output accumulator in C layout: oacc[t][r] = O[row r+8*hi][d = t*16+lo]
    v8f oacc[8] = {};
    float rowm[8], rowl[8];
#pragma unroll
    for (int r = 0; r < 8; ++r) { rowm[r] = -1.0e30f; rowl[r] = 0.0f; }

    // K loader: 256 threads = 64 rows x (4 threads x 32 cols)
    const int kRow = tid >> 2;
    const int kCol = (tid & 3) * 32;
    // V loader: 256 threads = 32 row-pairs x (8 threads x 16 cols)
    const int vRow = (tid >> 3) * 2;        // even key index
    const int vCol = (tid & 7) * 16;

#pragma unroll 1
    for (int kt = 0; kt < NKT; ++kt) {
        const int keyBase = kt * KT;

        // ---- cooperative stage of K (row-major) as f16 ----
        {
            const float4* ksrc = (const float4*)(K + ((size_t)bh * S_LEN + keyBase + kRow) * D_DIM + kCol);
            float kf[32];
#pragma unroll
            for (int j = 0; j < 8; ++j) {
                float4 t4 = ksrc[j];
                kf[4 * j + 0] = t4.x; kf[4 * j + 1] = t4.y;
                kf[4 * j + 2] = t4.z; kf[4 * j + 3] = t4.w;
            }
#pragma unroll
            for (int c8 = 0; c8 < 4; ++c8) {
                h8 tmp;
#pragma unroll
                for (int j = 0; j < 8; ++j) tmp[j] = (_Float16)kf[c8 * 8 + j];
                *(h8*)&kT[kRow][kCol + c8 * 8] = tmp;   // ds_store_b128
            }
        }
        // ---- cooperative stage of V transposed ([d][key]) as f16, 2 keys packed per b32 ----
        {
            const float4* v0 = (const float4*)(V + ((size_t)bh * S_LEN + keyBase + vRow) * D_DIM + vCol);
            const float4* v1 = (const float4*)(V + ((size_t)bh * S_LEN + keyBase + vRow + 1) * D_DIM + vCol);
            float va[16], vb[16];
#pragma unroll
            for (int j = 0; j < 4; ++j) {
                float4 a4 = v0[j], b4 = v1[j];
                va[4 * j + 0] = a4.x; va[4 * j + 1] = a4.y; va[4 * j + 2] = a4.z; va[4 * j + 3] = a4.w;
                vb[4 * j + 0] = b4.x; vb[4 * j + 1] = b4.y; vb[4 * j + 2] = b4.z; vb[4 * j + 3] = b4.w;
            }
#pragma unroll
            for (int j = 0; j < 16; ++j) {
                union { _Float16 h[2]; unsigned u; } pk;
                pk.h[0] = (_Float16)va[j];
                pk.h[1] = (_Float16)vb[j];
                *(unsigned*)&vT[vCol + j][vRow] = pk.u;  // ds_store_b32
            }
        }
        __syncthreads();

        // ---- scores: four 16x16 C tiles covering 64 keys, K-dim 128 ----
        v8f c[4] = {};
#pragma unroll
        for (int dc = 0; dc < 4; ++dc) {
#pragma unroll
            for (int g = 0; g < 4; ++g) {
                // B layout (16-bit B 32x16): lane n = key lo, halves 0..7 -> k = 8*hi + j, halves 8..15 -> k = 16 + 8*hi + j
                const h8* kp0 = (const h8*)&kT[16 * g + lo][dc * 32 + 8 * hi];
                const h8* kp1 = (const h8*)&kT[16 * g + lo][dc * 32 + 16 + 8 * hi];
                v16h b = __builtin_shufflevector(kp0[0], kp1[0],
                            0,1,2,3,4,5,6,7,8,9,10,11,12,13,14,15);
                c[g] = __builtin_amdgcn_wmma_f32_16x16x32_f16(false, qa[dc], false, b,
                                                              (short)0, c[g], false, false);
            }
        }

        // ---- online softmax (rows in VGPR index r, keys across 16 lanes x 4 tiles) ----
#pragma unroll
        for (int r = 0; r < 8; ++r) {
            float s0 = c[0][r], s1 = c[1][r], s2 = c[2][r], s3 = c[3][r];
            // reference: exact zeros in score matrix -> -1e9 before softmax
            s0 = (s0 == 0.0f) ? NEG_BIG : s0;
            s1 = (s1 == 0.0f) ? NEG_BIG : s1;
            s2 = (s2 == 0.0f) ? NEG_BIG : s2;
            s3 = (s3 == 0.0f) ? NEG_BIG : s3;
            float mx = fmaxf(fmaxf(s0, s1), fmaxf(s2, s3));
            mx = fmaxf(mx, __shfl_xor(mx, 1, 32));
            mx = fmaxf(mx, __shfl_xor(mx, 2, 32));
            mx = fmaxf(mx, __shfl_xor(mx, 4, 32));
            mx = fmaxf(mx, __shfl_xor(mx, 8, 32));
            const float mnew = fmaxf(rowm[r], mx);
            const float corr = __expf(rowm[r] - mnew);
            const float e0 = __expf(s0 - mnew);
            const float e1 = __expf(s1 - mnew);
            const float e2 = __expf(s2 - mnew);
            const float e3 = __expf(s3 - mnew);
            rowm[r] = mnew;
            float ps = (e0 + e1) + (e2 + e3);
            ps += __shfl_xor(ps, 1, 32);
            ps += __shfl_xor(ps, 2, 32);
            ps += __shfl_xor(ps, 4, 32);
            ps += __shfl_xor(ps, 8, 32);
            rowl[r] = rowl[r] * corr + ps;
#pragma unroll
            for (int t = 0; t < 8; ++t) oacc[t][r] *= corr;
            const int m = r + 8 * hi;
            pS[wave][m][lo]      = (_Float16)e0;
            pS[wave][m][16 + lo] = (_Float16)e1;
            pS[wave][m][32 + lo] = (_Float16)e2;
            pS[wave][m][48 + lo] = (_Float16)e3;
        }

        // intra-wave LDS exchange: keep the compiler from hoisting the reads;
        // DS ops from one wave execute in order, so no workgroup barrier needed.
        __builtin_amdgcn_wave_barrier();

        // ---- P (16x64) x V (64x128): 16 WMMAs accumulating into oacc ----
        const h8 pa0lo = *(const h8*)&pS[wave][lo][8 * hi];        // k = 8*hi + j
        const h8 pa0hi = *(const h8*)&pS[wave][lo][16 + 8 * hi];   // k = 16 + 8*hi + j
        const h8 pa1lo = *(const h8*)&pS[wave][lo][32 + 8 * hi];   // k = 32 + ...
        const h8 pa1hi = *(const h8*)&pS[wave][lo][48 + 8 * hi];
        v16h pa0 = __builtin_shufflevector(pa0lo, pa0hi,
                    0,1,2,3,4,5,6,7,8,9,10,11,12,13,14,15);
        v16h pa1 = __builtin_shufflevector(pa1lo, pa1hi,
                    0,1,2,3,4,5,6,7,8,9,10,11,12,13,14,15);
#pragma unroll
        for (int t = 0; t < 8; ++t) {
            // B operand: lane n = d col (t*16+lo), halves j<8 -> key = koff + 8*hi + j, j>=8 -> koff + 16 + 8*hi + (j-8)
            const h8* vp0 = (const h8*)&vT[t * 16 + lo][8 * hi];
            const h8* vp1 = (const h8*)&vT[t * 16 + lo][32 + 8 * hi];
            v16h bv0 = __builtin_shufflevector(vp0[0], vp0[2],
                        0,1,2,3,4,5,6,7,8,9,10,11,12,13,14,15);
            v16h bv1 = __builtin_shufflevector(vp1[0], vp1[2],
                        0,1,2,3,4,5,6,7,8,9,10,11,12,13,14,15);
            oacc[t] = __builtin_amdgcn_wmma_f32_16x16x32_f16(false, pa0, false, bv0,
                                                             (short)0, oacc[t], false, false);
            oacc[t] = __builtin_amdgcn_wmma_f32_16x16x32_f16(false, pa1, false, bv1,
                                                             (short)0, oacc[t], false, false);
        }
        __syncthreads();
    }

    // ---- epilogue: normalize and store fp32, coalesced along d ----
#pragma unroll
    for (int r = 0; r < 8; ++r) {
        const int m = r + 8 * hi;
        const float inv = 1.0f / rowl[r];
        float* orow = O + ((size_t)bh * S_LEN + qBase + m) * D_DIM;
#pragma unroll
        for (int t = 0; t < 8; ++t) {
            orow[t * 16 + lo] = oacc[t][r] * inv;
        }
    }
}

extern "C" void kernel_launch(void* const* d_in, const int* in_sizes, int n_in,
                              void* d_out, int out_size, void* d_ws, size_t ws_size,
                              hipStream_t stream) {
    (void)n_in; (void)out_size; (void)d_ws; (void)ws_size;
    const float* q = (const float*)d_in[0];
    const float* k = (const float*)d_in[1];
    const float* v = (const float*)d_in[2];
    float* o = (float*)d_out;
    const int bh = in_sizes[0] / (S_LEN * D_DIM);   // B*H = 64
    dim3 grid(S_LEN / QTILE, bh);
    dim3 block(256);
    fattn_f16wmma_kernel<<<grid, block, 0, stream>>>(q, k, v, o);
}
